// Net_86912958202606
// MI455X (gfx1250) — compile-verified, matrix operand order checked by reading, not run on hardware
//
#include <hip/hip_runtime.h>
#include <hip/hip_bf16.h>
#include <stddef.h>

// ---------------------------------------------------------------------------
// MI455X (gfx1250) fused 2-layer LSTM + projection.
//   B=256, S=512, H=256, E=8, V=90.
// Persistent kernel: 16 WGs (one per 16-row batch tile) x 512 threads
// (16 wave32s; wave w owns hidden tile w). All matmuls via
// v_wmma_f32_16x16x32_bf16 (f32 accumulate). State lives in registers/LDS;
// only logits hit HBM in steady state; packed weights stream from L2.
// ---------------------------------------------------------------------------

typedef __attribute__((ext_vector_type(16))) __bf16 v16bf;
typedef __attribute__((ext_vector_type(8)))  float  v8f;

#define HID   256
#define GATES 1024   // 4*HID
#define SEQ   512
#define BATCH 256

// workspace layout (bytes); all fragment bases 32B aligned
#define WHH0P_OFF 0u                     // 64 ntile * 8 kf * 512 bf16  = 512 KB
#define WC1P_OFF  524288u                // 64 ntile * 16 kf * 512 bf16 = 1 MB
#define WFCP_OFF  1572864u               // 6 ntile * 8 kf * 512 bf16   = 48 KB
#define T0_OFF    1622016u               // 90*1024 f32                 = 360 KB
#define B1_OFF    1990656u               // 1024 f32

// Fragment element mapping for 16-bit WMMA A/B operands (ISA 7.12.2):
// within a 32(K)x16(MN) fragment, lane holds mn = lane%16, and element e of
// its 16 packed bf16 maps to k = (e/8)*16 + (lane/16)*8 + (e%8).
__device__ __forceinline__ int kmap(int lane, int e) {
  return ((e >> 3) << 4) | ((lane >> 4) << 3) | (e & 7);
}

__device__ __forceinline__ float fsig(float z) {
  return 1.0f / (1.0f + __expf(-z));
}

// ---------------------------- packing kernels ------------------------------

__global__ void pack_whh0_k(const float* __restrict__ Whh0,
                            unsigned short* __restrict__ dstp) {
  __bf16* dst = reinterpret_cast<__bf16*>(dstp);
  int idx = blockIdx.x * 256 + threadIdx.x;          // < 64*8*512
  if (idx >= 64 * 8 * 512) return;
  int f = idx >> 9, r = idx & 511;
  int lane = r >> 4, e = r & 15;
  int nt = f >> 3, kf = f & 7;
  int n = nt * 16 + (lane & 15);
  int k = kf * 32 + kmap(lane, e);
  dst[idx] = (__bf16)Whh0[n * HID + k];
}

__global__ void pack_wc1_k(const float* __restrict__ Wih1,
                           const float* __restrict__ Whh1,
                           unsigned short* __restrict__ dstp) {
  __bf16* dst = reinterpret_cast<__bf16*>(dstp);
  int idx = blockIdx.x * 256 + threadIdx.x;          // < 64*16*512
  if (idx >= 64 * 16 * 512) return;
  int f = idx >> 9, r = idx & 511;
  int lane = r >> 4, e = r & 15;
  int nt = f >> 4, kf = f & 15;
  int n = nt * 16 + (lane & 15);
  int k = kf * 32 + kmap(lane, e);                   // 0..511: [Wih1 ; Whh1]
  float v = (k < HID) ? Wih1[n * HID + k] : Whh1[n * HID + (k - HID)];
  dst[idx] = (__bf16)v;
}

__global__ void pack_wfc_k(const float* __restrict__ Wfc,
                           unsigned short* __restrict__ dstp) {
  __bf16* dst = reinterpret_cast<__bf16*>(dstp);
  int idx = blockIdx.x * 256 + threadIdx.x;          // < 6*8*512
  if (idx >= 6 * 8 * 512) return;
  int f = idx >> 9, r = idx & 511;
  int lane = r >> 4, e = r & 15;
  int nt = f >> 3, kf = f & 7;
  int n = nt * 16 + (lane & 15);
  int k = kf * 32 + kmap(lane, e);
  float v = (n < 90) ? Wfc[n * HID + k] : 0.0f;
  dst[idx] = (__bf16)v;
}

// T0[v][n] = sum_e Wih0[n][e]*emb[v][e] + bih0[n] + bhh0[n]   (E=8 folded)
__global__ void build_t0_k(const float* __restrict__ Wih0,
                           const float* __restrict__ emb,
                           const float* __restrict__ bih0,
                           const float* __restrict__ bhh0,
                           float* __restrict__ t0) {
  int idx = blockIdx.x * 256 + threadIdx.x;          // < 90*1024
  if (idx >= 90 * GATES) return;
  int v = idx >> 10, n = idx & 1023;
  float acc = bih0[n] + bhh0[n];
#pragma unroll
  for (int e = 0; e < 8; ++e) acc += Wih0[n * 8 + e] * emb[v * 8 + e];
  t0[idx] = acc;
}

__global__ void build_b1_k(const float* __restrict__ bih1,
                           const float* __restrict__ bhh1,
                           float* __restrict__ b1) {
  int idx = blockIdx.x * 256 + threadIdx.x;
  if (idx < GATES) b1[idx] = bih1[idx] + bhh1[idx];
}

// --------------------------- fused LSTM kernel -----------------------------

__global__ __launch_bounds__(512, 1) void lstm2_fused_k(
    const int* __restrict__ x,              // [256][512]
    const float* __restrict__ t0,           // [90][1024]
    const float* __restrict__ b1,           // [1024]
    const float* __restrict__ bfc,          // [90]
    const unsigned short* __restrict__ whh0p_u,
    const unsigned short* __restrict__ wc1p_u,
    const unsigned short* __restrict__ wfcp_u,
    float* __restrict__ out)                // [256][512][90]
{
  // 16 K-fragments of [h0(s) ; h1(s-1)] in WMMA fragment layout, bf16. 16 KB.
  __shared__ __align__(32) __bf16 hcat[16 * 512];

  const int tid  = threadIdx.x;
  const int lane = tid & 31;
  const int w    = tid >> 5;       // wave id == hidden tile 0..15
  const int bt   = blockIdx.x;     // batch tile 0..15
  const int half = lane >> 4;
  const int n    = lane & 15;
  const int j    = w * 16 + n;     // hidden unit column owned by this lane

  for (int i = tid; i < 16 * 512; i += 512) hcat[i] = (__bf16)0.0f;

  float c0[8], c1[8];
#pragma unroll
  for (int v = 0; v < 8; ++v) { c0[v] = 0.0f; c1[v] = 0.0f; }

  // per-lane constants (invariant over the sequence)
  const float b1i = b1[j], b1f = b1[HID + j], b1g = b1[2 * HID + j],
              b1o = b1[3 * HID + j];
  const int fccol = w * 16 + n;
  float fcb = 0.0f;
  if (w < 6 && fccol < 90) fcb = bfc[fccol];

  const int* xrow = x + (size_t)bt * 16 * SEQ;   // rows bt*16 .. bt*16+15

  // h -> hcat fragment coordinates (inverse of kmap) for this lane's column j
  const int kfh = j >> 5, kk = j & 31;
  const int eh  = ((kk >> 4) << 3) | (kk & 7);
  const int lfbase = ((kk >> 3) & 1) << 4;       // + m gives fragment lane

  const __bf16* wbase0 = reinterpret_cast<const __bf16*>(whh0p_u);
  const __bf16* wbase1 = reinterpret_cast<const __bf16*>(wc1p_u);
  const __bf16* wbasef = reinterpret_cast<const __bf16*>(wfcp_u);

  // zoff is always 0 but opaque to the compiler: re-clobbered every timestep
  // so the (loop-invariant) weight-fragment loads cannot be hoisted out of
  // the sequence loop and spilled to scratch. Adding an offset (instead of
  // laundering the pointer through an integer) preserves the global
  // address-space inference -> global_load_b128, not flat_load_b128, so the
  // weight stream only ticks LOADcnt and stays decoupled from DScnt.
  unsigned int zoff = 0;

  __syncthreads();

  for (int s = 0; s < SEQ; ++s) {
    asm volatile("" : "+s"(zoff));
    const __bf16* wp0 = wbase0 + zoff;
    const __bf16* wp1 = wbase1 + zoff;
    const __bf16* wpf = wbasef + zoff;

    // ---- layer 0 recurrent GEMM: gates0 = h0 @ Whh0^T (K=256) ----
    v8f acc[4] = {};
#pragma unroll
    for (int kf = 0; kf < 8; ++kf) {
      v16bf a = *(const v16bf*)(&hcat[kf * 512 + lane * 16]);
      v16bf bf0 = *(const v16bf*)(wp0 + ((size_t)((0 * 16 + w) * 8 + kf)) * 512 + lane * 16);
      v16bf bf1 = *(const v16bf*)(wp0 + ((size_t)((1 * 16 + w) * 8 + kf)) * 512 + lane * 16);
      v16bf bf2 = *(const v16bf*)(wp0 + ((size_t)((2 * 16 + w) * 8 + kf)) * 512 + lane * 16);
      v16bf bf3 = *(const v16bf*)(wp0 + ((size_t)((3 * 16 + w) * 8 + kf)) * 512 + lane * 16);
      acc[0] = __builtin_amdgcn_wmma_f32_16x16x32_bf16(false, a, false, bf0,
                                                       (short)0, acc[0], false, false);
      acc[1] = __builtin_amdgcn_wmma_f32_16x16x32_bf16(false, a, false, bf1,
                                                       (short)0, acc[1], false, false);
      acc[2] = __builtin_amdgcn_wmma_f32_16x16x32_bf16(false, a, false, bf2,
                                                       (short)0, acc[2], false, false);
      acc[3] = __builtin_amdgcn_wmma_f32_16x16x32_bf16(false, a, false, bf3,
                                                       (short)0, acc[3], false, false);
    }
    __syncthreads();   // all reads of hcat[0..7] done before EW0 rewrites

    // ---- layer 0 gates (in-register) + embedding-table gather ----
#pragma unroll
    for (int v = 0; v < 8; ++v) {
      int m  = v + 8 * half;
      int xv = xrow[m * SEQ + s];
      const float* t0r = t0 + (size_t)xv * GATES;
      float iv = fsig(acc[0][v] + t0r[j]);
      float fv = fsig(acc[1][v] + t0r[HID + j]);
      float gv = tanhf(acc[2][v] + t0r[2 * HID + j]);
      float ov = fsig(acc[3][v] + t0r[3 * HID + j]);
      c0[v] = fv * c0[v] + iv * gv;
      float hval = ov * tanhf(c0[v]);
      hcat[kfh * 512 + (m + lfbase) * 16 + eh] = (__bf16)hval;
    }
    __syncthreads();   // hcat[0..7] = h0(s)

    // ---- layer 1 GEMM: gates1 = [h0(s);h1(s-1)] @ [Wih1;Whh1]^T (K=512) ----
    v8f acc1[4] = {};
#pragma unroll
    for (int kf = 0; kf < 16; ++kf) {
      v16bf a = *(const v16bf*)(&hcat[kf * 512 + lane * 16]);
      v16bf bf0 = *(const v16bf*)(wp1 + ((size_t)((0 * 16 + w) * 16 + kf)) * 512 + lane * 16);
      v16bf bf1 = *(const v16bf*)(wp1 + ((size_t)((1 * 16 + w) * 16 + kf)) * 512 + lane * 16);
      v16bf bf2 = *(const v16bf*)(wp1 + ((size_t)((2 * 16 + w) * 16 + kf)) * 512 + lane * 16);
      v16bf bf3 = *(const v16bf*)(wp1 + ((size_t)((3 * 16 + w) * 16 + kf)) * 512 + lane * 16);
      acc1[0] = __builtin_amdgcn_wmma_f32_16x16x32_bf16(false, a, false, bf0,
                                                        (short)0, acc1[0], false, false);
      acc1[1] = __builtin_amdgcn_wmma_f32_16x16x32_bf16(false, a, false, bf1,
                                                        (short)0, acc1[1], false, false);
      acc1[2] = __builtin_amdgcn_wmma_f32_16x16x32_bf16(false, a, false, bf2,
                                                        (short)0, acc1[2], false, false);
      acc1[3] = __builtin_amdgcn_wmma_f32_16x16x32_bf16(false, a, false, bf3,
                                                        (short)0, acc1[3], false, false);
    }
    __syncthreads();   // all reads of hcat done before EW1 rewrites 8..15

    // ---- layer 1 gates (in-register) ----
#pragma unroll
    for (int v = 0; v < 8; ++v) {
      int m = v + 8 * half;
      float iv = fsig(acc1[0][v] + b1i);
      float fv = fsig(acc1[1][v] + b1f);
      float gv = tanhf(acc1[2][v] + b1g);
      float ov = fsig(acc1[3][v] + b1o);
      c1[v] = fv * c1[v] + iv * gv;
      float hval = ov * tanhf(c1[v]);
      hcat[(8 + kfh) * 512 + (m + lfbase) * 16 + eh] = (__bf16)hval;
    }
    __syncthreads();   // hcat[8..15] = h1(s)

    // ---- fused projection: logits(s) = h1(s) @ Wfc^T + bfc (waves 0..5) ----
    if (w < 6) {
      v8f accf = {};
#pragma unroll
      for (int kf = 0; kf < 8; ++kf) {
        v16bf a   = *(const v16bf*)(&hcat[(8 + kf) * 512 + lane * 16]);
        v16bf bfr = *(const v16bf*)(wpf + ((size_t)(w * 8 + kf)) * 512 +
                                    lane * 16);
        accf = __builtin_amdgcn_wmma_f32_16x16x32_bf16(
            false, a, false, bfr, (short)0, accf, false, false);
      }
      if (fccol < 90) {
#pragma unroll
        for (int v = 0; v < 8; ++v) {
          int m = v + 8 * half;
          out[((size_t)(bt * 16 + m) * SEQ + s) * 90 + fccol] = accf[v] + fcb;
        }
      }
    }
    // next-iteration GEMM0 touches only hcat[0..7]; FC reads hcat[8..15],
    // which is rewritten only after the next GEMM1 barrier -> no extra sync.
  }
}

// ------------------------------- launcher ----------------------------------

extern "C" void kernel_launch(void* const* d_in, const int* in_sizes, int n_in,
                              void* d_out, int out_size, void* d_ws,
                              size_t ws_size, hipStream_t stream) {
  (void)in_sizes; (void)n_in; (void)out_size; (void)ws_size;
  const int*   x    = (const int*)d_in[0];
  const float* emb  = (const float*)d_in[1];
  const float* Wih0 = (const float*)d_in[2];
  const float* Whh0 = (const float*)d_in[3];
  const float* bih0 = (const float*)d_in[4];
  const float* bhh0 = (const float*)d_in[5];
  const float* Wih1 = (const float*)d_in[6];
  const float* Whh1 = (const float*)d_in[7];
  const float* bih1 = (const float*)d_in[8];
  const float* bhh1 = (const float*)d_in[9];
  const float* Wfc  = (const float*)d_in[10];
  const float* bfc  = (const float*)d_in[11];
  float* out = (float*)d_out;

  char* ws = (char*)d_ws;
  unsigned short* whh0p = (unsigned short*)(ws + WHH0P_OFF);
  unsigned short* wc1p  = (unsigned short*)(ws + WC1P_OFF);
  unsigned short* wfcp  = (unsigned short*)(ws + WFCP_OFF);
  float*          t0    = (float*)(ws + T0_OFF);
  float*          b1    = (float*)(ws + B1_OFF);

  pack_whh0_k<<<(64 * 8 * 512) / 256, 256, 0, stream>>>(Whh0, whh0p);
  pack_wc1_k<<<(64 * 16 * 512) / 256, 256, 0, stream>>>(Wih1, Whh1, wc1p);
  pack_wfc_k<<<(6 * 8 * 512) / 256, 256, 0, stream>>>(Wfc, wfcp);
  build_t0_k<<<(90 * GATES) / 256, 256, 0, stream>>>(Wih0, emb, bih0, bhh0, t0);
  build_b1_k<<<GATES / 256, 256, 0, stream>>>(bih1, bhh1, b1);

  lstm2_fused_k<<<16, 512, 0, stream>>>(x, t0, b1, bfc, whh0p, wc1p, wfcp, out);
}
